// TimeMix_7_5463198400615
// MI455X (gfx1250) — matrix operand order repaired and checked by reading, not compiled, not run on hardware
//
#include <hip/hip_runtime.h>

// RWKV-7 TimeMix for MI455X (gfx1250, wave32, WMMA).
// Pipeline:
//   K0  x4 : W (fp32, KxN) -> Wt (bf16, NxK)  [WMMA B-operand layout prep]
//   K1     : time-shift + low-rank token-mix -> six branch activations (bf16)
//   K2  x3 : bf16 WMMA GEMM  r = xr@W_r, k = xk@W_k, v = xv@W_v (fp32 out)
//   K3     : LoRA branches (decay / kkk / aaa / gate), kk-norm, k-scale, wd
//   K4     : sequential RWKV-7 state scan (32 blocks, state in registers,
//            global loads software-pipelined one timestep ahead)
//   K5     : per-head GroupNorm + bonus + gate multiply -> bf16
//   K6     : bf16 WMMA GEMM  out = (y*g)@W_o  -> d_out (fp32)
// GEMM: one wave computes a 32(M)x64(N) tile; per K-step all 12 b128 loads
// (2 A frags + 4 B frags) issue as one clause into distinct registers, then
// 8 v_wmma drain with staggered s_wait_loadcnt.
// Workspace: ~201 MB of d_ws, all buffers fully rewritten each launch.

static constexpr int kH  = 16;
static constexpr int kN  = 64;
static constexpr int kC  = 1024;
static constexpr int kT  = 2048;
static constexpr int kB  = 2;
static constexpr int kBT = kB * kT;                     // 4096 tokens
static constexpr long long kBTC = (long long)kBT * kC;  // 4,194,304

typedef __attribute__((ext_vector_type(16))) __bf16 v16bf;
typedef __attribute__((ext_vector_type(8)))  __bf16 v8bf;
typedef __attribute__((ext_vector_type(8)))  float  v8f;

__device__ __forceinline__ __bf16 f2bf(float f) {
    unsigned int u = __builtin_bit_cast(unsigned int, f);
    unsigned int r = u + 0x7FFFu + ((u >> 16) & 1u);   // RNE
    unsigned short h = (unsigned short)(r >> 16);
    return __builtin_bit_cast(__bf16, h);
}
__device__ __forceinline__ float bf2f(__bf16 h) {
    unsigned int u = ((unsigned int)__builtin_bit_cast(unsigned short, h)) << 16;
    return __builtin_bit_cast(float, u);
}
__device__ __forceinline__ v16bf cat8(v8bf lo, v8bf hi) {
    return __builtin_shufflevector(lo, hi,
        0, 1, 2, 3, 4, 5, 6, 7, 8, 9, 10, 11, 12, 13, 14, 15);
}

// ---------------- K0: weight transpose + bf16 convert: Wt[n][k] = W[k][n] ----
__global__ __launch_bounds__(256) void wt_bf16_kernel(
    const float* __restrict__ W, __bf16* __restrict__ Wt) {
    int idx = blockIdx.x * 256 + threadIdx.x;           // over C*C
    int n = idx >> 10, k = idx & (kC - 1);
    Wt[idx] = f2bf(W[(size_t)k * kC + n]);
}

// ---------------- K1: time-shift + token-mix interpolation ------------------
__global__ __launch_bounds__(256) void mix_kernel(
    const float* __restrict__ x,
    const float* __restrict__ maa_x, const float* __restrict__ maa_r,
    const float* __restrict__ maa_w, const float* __restrict__ maa_k,
    const float* __restrict__ maa_v, const float* __restrict__ maa_a,
    const float* __restrict__ maa_g,
    const float* __restrict__ w1,   // (C,192)
    const float* __restrict__ w2,   // (6,32,C)
    __bf16* __restrict__ xbr, __bf16* __restrict__ xbw, __bf16* __restrict__ xbk,
    __bf16* __restrict__ xbv, __bf16* __restrict__ xba, __bf16* __restrict__ xbg)
{
    __shared__ float sX[kC], sXX[kC], sXM[kC], sMIX[192];
    const int bt = blockIdx.x;
    const int t  = bt & (kT - 1);
    const float* xc = x + (size_t)bt * kC;
    const float* xp = xc - kC;                          // valid only if t>0
    for (int c = threadIdx.x; c < kC; c += 256) {
        float cur = xc[c];
        float xx  = (t > 0 ? xp[c] : 0.0f) - cur;       // xx[t] = x[t-1]-x[t]
        sX[c] = cur; sXX[c] = xx;
        sXM[c] = cur + xx * maa_x[c];
    }
    __syncthreads();
    if (threadIdx.x < 192) {
        const int o = threadIdx.x;
        float s = 0.0f;
        for (int c = 0; c < kC; ++c) s += sXM[c] * w1[c * 192 + o];
        sMIX[o] = tanhf(s);
    }
    __syncthreads();
    for (int c = threadIdx.x; c < kC; c += 256) {
        float m[6];
        #pragma unroll
        for (int sb = 0; sb < 6; ++sb) {
            float acc = 0.0f;
            const float* w2p = w2 + (size_t)(sb * 32) * kC + c;
            #pragma unroll 8
            for (int d = 0; d < 32; ++d) acc += sMIX[sb * 32 + d] * w2p[(size_t)d * kC];
            m[sb] = acc;
        }
        float cur = sX[c], xx = sXX[c];
        size_t idx = (size_t)bt * kC + c;
        xbr[idx] = f2bf(cur + xx * (maa_r[c] + m[0]));
        xbw[idx] = f2bf(cur + xx * (maa_w[c] + m[1]));
        xbk[idx] = f2bf(cur + xx * (maa_k[c] + m[2]));
        xbv[idx] = f2bf(cur + xx * (maa_v[c] + m[3]));
        xba[idx] = f2bf(cur + xx * (maa_a[c] + m[4]));
        xbg[idx] = f2bf(cur + xx * (maa_g[c] + m[5]));
    }
}

// ---------------- K2/K6: bf16 WMMA GEMM  Out[M,C] = A[M,C] @ Wt[C,C]^T ------
// A row-major (M x K), Wt row-major (N x K) i.e. pre-transposed weight.
// One wave -> 32(M) x 64(N) tile (2 M-tiles x 4 N-tiles). Per K-step the
// 2 A fragments + 4 B fragments are preloaded into distinct registers
// (one 12-load clause), then 8 v_wmma drain with staggered waits.
__global__ __launch_bounds__(256) void gemm_bf16_wmma_kernel(
    const __bf16* __restrict__ A, const __bf16* __restrict__ Wt,
    float* __restrict__ Out)
{
    const int wv   = threadIdx.x >> 5;                  // wave in block 0..7
    const int lid  = threadIdx.x & 31;
    const int half = lid >> 4;                          // lane group 0/1
    const int lr   = lid & 15;
    const int m0   = blockIdx.x * 32;                   // 2 M-tiles per wave
    const int nb   = (blockIdx.y * 8 + wv) * 64;        // N strip base
    const __bf16* arow0 = A + (size_t)(m0 + lr) * kC + half * 8;
    const __bf16* arow1 = arow0 + (size_t)16 * kC;
    const __bf16* brow0 = Wt + (size_t)(nb + lr) * kC + half * 8;

    v8f acc[2][4] = {};
    for (int kb = 0; kb < kC; kb += 32) {
        // A fragments: halves [0..7]=A[m][kb+koff..+7], [8..15]=A[m][kb+koff+16..+23]
        v16bf af0 = cat8(*(const v8bf*)(arow0 + kb), *(const v8bf*)(arow0 + kb + 16));
        v16bf af1 = cat8(*(const v8bf*)(arow1 + kb), *(const v8bf*)(arow1 + kb + 16));
        // B fragments: column n = Wt row; [0..7]=K kb+8*half..+7, [8..15]=K kb+16+8*half..+7
        v16bf bf_[4];
        #pragma unroll
        for (int nt = 0; nt < 4; ++nt) {
            const __bf16* brow = brow0 + (size_t)(nt * 16) * kC + kb;
            bf_[nt] = cat8(*(const v8bf*)brow, *(const v8bf*)(brow + 16));
        }
        #pragma unroll
        for (int nt = 0; nt < 4; ++nt) {
            acc[0][nt] = __builtin_amdgcn_wmma_f32_16x16x32_bf16(
                false, af0, false, bf_[nt], (short)0, acc[0][nt], false, false);
            acc[1][nt] = __builtin_amdgcn_wmma_f32_16x16x32_bf16(
                false, af1, false, bf_[nt], (short)0, acc[1][nt], false, false);
        }
    }
    #pragma unroll
    for (int mt = 0; mt < 2; ++mt) {
        #pragma unroll
        for (int nt = 0; nt < 4; ++nt) {
            const int col = nb + nt * 16 + lr;
            #pragma unroll
            for (int vv = 0; vv < 8; ++vv) {
                const int row = m0 + mt * 16 + half * 8 + vv;  // D: M = vv + 8*half
                Out[(size_t)row * kC + col] = acc[mt][nt][vv];
            }
        }
    }
}

// ---------------- K3: LoRA branches + kk-norm + decay + k-scale -------------
__global__ __launch_bounds__(256) void lora_kernel(
    const __bf16* __restrict__ xbw, const __bf16* __restrict__ xbk,
    const __bf16* __restrict__ xba, const __bf16* __restrict__ xbg,
    float* __restrict__ k,                       // in: raw k; out: k*exp(min(.5w,0))
    const float* __restrict__ tdecay, const float* __restrict__ taaaaa,
    const float* __restrict__ dw1, const float* __restrict__ dw2,
    const float* __restrict__ aw1, const float* __restrict__ aw2,
    const float* __restrict__ kw1, const float* __restrict__ kw2,
    const float* __restrict__ gw1, const float* __restrict__ gw2,
    float* __restrict__ wd, float* __restrict__ aa,
    float* __restrict__ bb, float* __restrict__ g)
{
    __shared__ float sXW[kC], sXK[kC], sXA[kC], sXG[kC], sKK[kC];
    __shared__ float sH[128], sRed[256];
    const int tid = threadIdx.x;
    const size_t base = (size_t)blockIdx.x * kC;
    for (int c = tid; c < kC; c += 256) {
        sXW[c] = bf2f(xbw[base + c]); sXK[c] = bf2f(xbk[base + c]);
        sXA[c] = bf2f(xba[base + c]); sXG[c] = bf2f(xbg[base + c]);
    }
    __syncthreads();
    // ---- kkk branch: kk = k + tanh(xk@kw1)@kw2, normalized over C ----
    if (tid < 64) {
        float s = 0.0f;
        for (int c = 0; c < kC; ++c) s += sXK[c] * kw1[c * 64 + tid];
        sH[tid] = tanhf(s);
    }
    __syncthreads();
    for (int c = tid; c < kC; c += 256) {
        float acc = 0.0f;
        #pragma unroll 8
        for (int d = 0; d < 64; ++d) acc += sH[d] * kw2[(size_t)d * kC + c];
        sKK[c] = k[base + c] + acc;
    }
    __syncthreads();
    float ss = 0.0f;
    for (int c = tid; c < kC; c += 256) ss += sKK[c] * sKK[c];
    sRed[tid] = ss;
    __syncthreads();
    for (int off = 128; off > 0; off >>= 1) {
        if (tid < off) sRed[tid] += sRed[tid + off];
        __syncthreads();
    }
    const float rnorm = 1.0f / fmaxf(sqrtf(sRed[0]), 1e-12f);
    __syncthreads();
    // ---- aaa branch: a = 2*sigmoid(taaaaa + (xa@aw1)@aw2) ----
    if (tid < 64) {
        float s = 0.0f;
        for (int c = 0; c < kC; ++c) s += sXA[c] * aw1[c * 64 + tid];
        sH[tid] = s;                                    // no tanh here
    }
    __syncthreads();
    for (int c = tid; c < kC; c += 256) {
        float av = 0.0f;
        #pragma unroll 8
        for (int d = 0; d < 64; ++d) av += sH[d] * aw2[(size_t)d * kC + c];
        av = 2.0f / (1.0f + expf(-(taaaaa[c] + av)));
        float kkn = sKK[c] * rnorm;
        aa[base + c] = -kkn;                            // a-vector of the scan
        bb[base + c] = kkn * av;                        // b-vector of the scan
    }
    __syncthreads();
    // ---- decay branch: w = -softplus(-(tdecay + tanh(xw@dw1)@dw2)) - 0.5 ----
    if (tid < 64) {
        float s = 0.0f;
        for (int c = 0; c < kC; ++c) s += sXW[c] * dw1[c * 64 + tid];
        sH[tid] = tanhf(s);
    }
    __syncthreads();
    for (int c = tid; c < kC; c += 256) {
        float dec = 0.0f;
        #pragma unroll 8
        for (int d = 0; d < 64; ++d) dec += sH[d] * dw2[(size_t)d * kC + c];
        float z  = tdecay[c] + dec;
        float u  = -z;
        float sp = (u > 20.0f) ? u : log1pf(expf(u));
        float wv = -sp - 0.5f;
        wd[base + c] = expf(-expf(wv));
        float kv = k[base + c];
        k[base + c] = kv * expf(fminf(0.5f * wv, 0.0f));
    }
    __syncthreads();
    // ---- gate branch: g = tanh(xg@gw1)@gw2  (rank 128) ----
    if (tid < 128) {
        float s = 0.0f;
        for (int c = 0; c < kC; ++c) s += sXG[c] * gw1[c * 128 + tid];
        sH[tid] = tanhf(s);
    }
    __syncthreads();
    for (int c = tid; c < kC; c += 256) {
        float gv = 0.0f;
        #pragma unroll 8
        for (int d = 0; d < 128; ++d) gv += sH[d] * gw2[(size_t)d * kC + c];
        g[base + c] = gv;
    }
}

// ---------------- K4: sequential RWKV-7 scan --------------------------------
// One block per (b,h). thread = (row i)*4 + quarter q; S[i][16q..16q+15] in regs.
// Global loads for step t+1 are issued right after step t is published to LDS,
// so the memory round-trip overlaps a full step of compute.
__global__ __launch_bounds__(256) void wkv7_kernel(
    const float* __restrict__ r,  const float* __restrict__ wd,
    const float* __restrict__ k,  const float* __restrict__ v,
    const float* __restrict__ aa, const float* __restrict__ bb,
    float* __restrict__ y)
{
    __shared__ float lr_[kN], lw_[kN], lk_[kN], lv_[kN], la_[kN], lb_[kN];
    const int b = blockIdx.x >> 4, h = blockIdx.x & 15;
    const int tid = threadIdx.x;
    const int i = tid >> 2, q = tid & 3, j0 = q * 16;
    const bool ld = (tid < kN);
    float S[16];
    #pragma unroll
    for (int jj = 0; jj < 16; ++jj) S[jj] = 0.0f;

    size_t base = (size_t)(b * kT) * kC + h * kN;       // token t=0
    float pr = 0, pw = 0, pk = 0, pv = 0, pa = 0, pb = 0;
    if (ld) {
        pr = r[base + tid];  pw = wd[base + tid]; pk = k[base + tid];
        pv = v[base + tid];  pa = aa[base + tid]; pb = bb[base + tid];
    }
    for (int t = 0; t < kT; ++t) {
        if (ld) {
            lr_[tid] = pr; lw_[tid] = pw; lk_[tid] = pk;
            lv_[tid] = pv; la_[tid] = pa; lb_[tid] = pb;
        }
        __syncthreads();
        if (ld && (t + 1 < kT)) {                       // prefetch next token
            const size_t nxt = base + kC;
            pr = r[nxt + tid];  pw = wd[nxt + tid]; pk = k[nxt + tid];
            pv = v[nxt + tid];  pa = aa[nxt + tid]; pb = bb[nxt + tid];
        }
        float sa = 0.0f;
        #pragma unroll
        for (int jj = 0; jj < 16; ++jj) sa += S[jj] * la_[j0 + jj];
        sa += __shfl_xor(sa, 1, 32);                    // reduce over q (4 lanes)
        sa += __shfl_xor(sa, 2, 32);
        const float vi = lv_[i];
        float yp = 0.0f;
        #pragma unroll
        for (int jj = 0; jj < 16; ++jj) {
            const int j = j0 + jj;
            float s2 = S[jj] * lw_[j] + sa * lb_[j] + vi * lk_[j];
            S[jj] = s2;
            yp += s2 * lr_[j];
        }
        yp += __shfl_xor(yp, 1, 32);
        yp += __shfl_xor(yp, 2, 32);
        if (q == 0) y[base + i] = yp;
        base += kC;
        __syncthreads();
    }
}

// ---------------- K5: GroupNorm + bonus + gate multiply -> bf16 -------------
__global__ __launch_bounds__(256) void post_kernel(
    const float* __restrict__ y, const float* __restrict__ r,
    const float* __restrict__ k, const float* __restrict__ v,
    const float* __restrict__ g, const float* __restrict__ faaaa,
    const float* __restrict__ lnw, const float* __restrict__ lnb,
    __bf16* __restrict__ ygb)
{
    __shared__ float sY[kC], sMean[kH], sRstd[kH], sBon[kH];
    const size_t base = (size_t)blockIdx.x * kC;
    for (int c = threadIdx.x; c < kC; c += 256) sY[c] = y[base + c];
    __syncthreads();
    if (threadIdx.x < kH) {
        const int h = threadIdx.x;
        float m = 0.0f, m2 = 0.0f, bon = 0.0f;
        for (int n = 0; n < kN; ++n) {
            float yv = sY[h * kN + n];
            m += yv; m2 += yv * yv;
            bon += r[base + h * kN + n] * k[base + h * kN + n] * faaaa[h * kN + n];
        }
        m *= (1.0f / kN);
        m2 = m2 * (1.0f / kN) - m * m;                  // biased variance
        sMean[h] = m;
        sRstd[h] = rsqrtf(m2 + 6.4e-4f);                // LN_EPS = 1e-5*64
        sBon[h]  = bon;
    }
    __syncthreads();
    for (int c = threadIdx.x; c < kC; c += 256) {
        const int h = c >> 6;
        float gn = (sY[c] - sMean[h]) * sRstd[h] * lnw[c] + lnb[c];
        float yb = gn + sBon[h] * v[base + c];
        ygb[base + c] = f2bf(yb * g[base + c]);
    }
}

// ---------------- host-side orchestration -----------------------------------
extern "C" void kernel_launch(void* const* d_in, const int* in_sizes, int n_in,
                              void* d_out, int out_size, void* d_ws, size_t ws_size,
                              hipStream_t stream) {
    (void)in_sizes; (void)n_in; (void)out_size; (void)ws_size;
    const float* x      = (const float*)d_in[0];
    const float* maa_x  = (const float*)d_in[1];
    const float* maa_r  = (const float*)d_in[2];
    const float* maa_w  = (const float*)d_in[3];
    const float* maa_k  = (const float*)d_in[4];
    const float* maa_v  = (const float*)d_in[5];
    const float* maa_a  = (const float*)d_in[6];
    const float* maa_g  = (const float*)d_in[7];
    const float* tdecay = (const float*)d_in[8];
    const float* faaaa  = (const float*)d_in[9];
    const float* taaaaa = (const float*)d_in[10];
    const float* mw1    = (const float*)d_in[11];
    const float* mw2    = (const float*)d_in[12];
    const float* dw1    = (const float*)d_in[13];
    const float* dw2    = (const float*)d_in[14];
    const float* aw1    = (const float*)d_in[15];
    const float* aw2    = (const float*)d_in[16];
    const float* kw1    = (const float*)d_in[17];
    const float* kw2    = (const float*)d_in[18];
    const float* gw1    = (const float*)d_in[19];
    const float* gw2    = (const float*)d_in[20];
    const float* W_r    = (const float*)d_in[21];
    const float* W_k    = (const float*)d_in[22];
    const float* W_v    = (const float*)d_in[23];
    const float* W_o    = (const float*)d_in[24];
    const float* lnw    = (const float*)d_in[25];
    const float* lnb    = (const float*)d_in[26];

    // workspace carve-up (~201 MB total)
    char* ws = (char*)d_ws;
    size_t off = 0;
    auto carve = [&](size_t bytes) { void* p = ws + off; off += (bytes + 255) & ~(size_t)255; return p; };
    __bf16* xbr = (__bf16*)carve(kBTC * 2);
    __bf16* xbw = (__bf16*)carve(kBTC * 2);
    __bf16* xbk = (__bf16*)carve(kBTC * 2);
    __bf16* xbv = (__bf16*)carve(kBTC * 2);
    __bf16* xba = (__bf16*)carve(kBTC * 2);
    __bf16* xbg = (__bf16*)carve(kBTC * 2);
    __bf16* ygb = (__bf16*)carve(kBTC * 2);
    __bf16* WtR = (__bf16*)carve((size_t)kC * kC * 2);
    __bf16* WtK = (__bf16*)carve((size_t)kC * kC * 2);
    __bf16* WtV = (__bf16*)carve((size_t)kC * kC * 2);
    __bf16* WtO = (__bf16*)carve((size_t)kC * kC * 2);
    float* rbuf  = (float*)carve(kBTC * 4);
    float* kbuf  = (float*)carve(kBTC * 4);
    float* vbuf  = (float*)carve(kBTC * 4);
    float* wdbuf = (float*)carve(kBTC * 4);
    float* aabuf = (float*)carve(kBTC * 4);
    float* bbbuf = (float*)carve(kBTC * 4);
    float* ybuf  = (float*)carve(kBTC * 4);
    float* gbuf  = (float*)carve(kBTC * 4);

    const dim3 blk(256);
    const int wtBlocks = (kC * kC) / 256;

    // K0: transpose+convert weights for the WMMA B-operand path
    wt_bf16_kernel<<<wtBlocks, blk, 0, stream>>>(W_r, WtR);
    wt_bf16_kernel<<<wtBlocks, blk, 0, stream>>>(W_k, WtK);
    wt_bf16_kernel<<<wtBlocks, blk, 0, stream>>>(W_v, WtV);
    wt_bf16_kernel<<<wtBlocks, blk, 0, stream>>>(W_o, WtO);

    // K1: token-mix
    mix_kernel<<<kBT, blk, 0, stream>>>(x, maa_x, maa_r, maa_w, maa_k, maa_v,
                                        maa_a, maa_g, mw1, mw2,
                                        xbr, xbw, xbk, xbv, xba, xbg);

    // K2: big projections via WMMA (grid: 128 M-slabs x 2 N-superstrips)
    const dim3 ggrid(kBT / 32, kC / 512);
    gemm_bf16_wmma_kernel<<<ggrid, blk, 0, stream>>>(xbr, WtR, rbuf);
    gemm_bf16_wmma_kernel<<<ggrid, blk, 0, stream>>>(xbk, WtK, kbuf);
    gemm_bf16_wmma_kernel<<<ggrid, blk, 0, stream>>>(xbv, WtV, vbuf);

    // K3: LoRA branches, kk-normalization, decay, k-scaling
    lora_kernel<<<kBT, blk, 0, stream>>>(xbw, xbk, xba, xbg, kbuf,
                                         tdecay, taaaaa, dw1, dw2, aw1, aw2,
                                         kw1, kw2, gw1, gw2,
                                         wdbuf, aabuf, bbbuf, gbuf);

    // K4: sequential scan (B*H = 32 independent states)
    wkv7_kernel<<<kB * kH, blk, 0, stream>>>(rbuf, wdbuf, kbuf, vbuf,
                                             aabuf, bbbuf, ybuf);

    // K5: GroupNorm + bonus + gate
    post_kernel<<<kBT, blk, 0, stream>>>(ybuf, rbuf, kbuf, vbuf, gbuf,
                                         faaaa, lnw, lnb, ygb);

    // K6: output projection via WMMA -> d_out (fp32)
    gemm_bf16_wmma_kernel<<<ggrid, blk, 0, stream>>>(ygb, WtO, (float*)d_out);
}
